// NETC_high_58901181497818
// MI455X (gfx1250) — matrix operand order repaired and checked by reading, not compiled, not run on hardware
//
#include <hip/hip_runtime.h>
#include <hip/hip_bf16.h>

typedef _Float16 v16h __attribute__((ext_vector_type(16)));
typedef _Float16 v2h  __attribute__((ext_vector_type(2)));
typedef float    v8f  __attribute__((ext_vector_type(8)));

#define TARGET_VAL 6.2562059f   // 0.62562059 * 10

// ---------------------------------------------------------------------------
// Kernel 1: compute dot_target = h2(target) . w3  (b3 cancels in u = f(x)-f(t))
// One block of 64 threads, exact fp32.
// ---------------------------------------------------------------------------
__global__ __launch_bounds__(64)
void netc_target_kernel(const float* __restrict__ W1, const float* __restrict__ b1,
                        const float* __restrict__ W2, const float* __restrict__ b2,
                        const float* __restrict__ W3, float* __restrict__ ws) {
    __shared__ float h1s[64];
    __shared__ float ps[64];
    int k = threadIdx.x;                       // 0..63
    float h = TARGET_VAL * W1[2 * k] + TARGET_VAL * W1[2 * k + 1] + b1[k];
    h1s[k] = fmaxf(h, 0.0f);
    __syncthreads();
    float s = b2[k];
    #pragma unroll 8
    for (int j = 0; j < 64; ++j) s = fmaf(h1s[j], W2[k * 64 + j], s);
    ps[k] = fmaxf(s, 0.0f) * W3[k];
    __syncthreads();
    if (k == 0) {
        float t = 0.0f;
        #pragma unroll 8
        for (int j = 0; j < 64; ++j) t += ps[j];
        ws[0] = t;
    }
}

// ---------------------------------------------------------------------------
// Kernel 2: fused MLP (WMMA f16) + Hill dynamics. One wave per 16-row tile.
// ---------------------------------------------------------------------------
__global__ __launch_bounds__(256)
void netc_main_kernel(const float* __restrict__ x,  const float* __restrict__ y,
                      const float* __restrict__ ex, const float* __restrict__ ey,
                      const float* __restrict__ W1, const float* __restrict__ b1,
                      const float* __restrict__ W2, const float* __restrict__ b2,
                      const float* __restrict__ W3, const float* __restrict__ ws,
                      float* __restrict__ out, int n, int nTiles) {
    const int lane    = threadIdx.x & 31;
    const int wid     = threadIdx.x >> 5;      // wave in block (8 waves/block)
    const int gwave   = blockIdx.x * 8 + wid;
    const int nWaves  = gridDim.x * 8;
    const int col     = lane & 15;             // N / M-row index within tile
    const int hiHalf  = (lane >= 16) ? 1 : 0;

    const float dotT = ws[0];                  // uniform scalar from kernel 1

    // ---- Layer-1 weights in A-operand order (packed f16 pairs), resident ----
    // A-operand (16-bit, 16x32): lane<16 holds K = {0..7, 16..23}+a*32,
    //                            lane>=16 holds K = {8..15, 24..31}+a*32.
    v2h w1xp[16], w1yp[16], b1p[16];
    #pragma unroll
    for (int a = 0; a < 2; ++a) {
        #pragma unroll
        for (int p = 0; p < 8; ++p) {
            int idx = a * 8 + p;
            int k0  = hiHalf * 8 + a * 32 + 2 * p + ((p >= 4) ? 8 : 0);
            w1xp[idx] = (v2h){(_Float16)W1[2 * k0],     (_Float16)W1[2 * k0 + 2]};
            w1yp[idx] = (v2h){(_Float16)W1[2 * k0 + 1], (_Float16)W1[2 * k0 + 3]};
            b1p[idx]  = (v2h){(_Float16)b1[k0],         (_Float16)b1[k0 + 1]};
        }
    }

    // ---- W2 as resident B-operands: B[k][n] = W2[n][k] ----
    // B-operand (16-bit, 32x16): lanes 0-15 hold K={0..7,16..23}, lanes 16-31 K={8..15,24..31}.
    v16h Bop[8];
    #pragma unroll
    for (int nb = 0; nb < 4; ++nb) {
        #pragma unroll
        for (int kb = 0; kb < 2; ++kb) {
            v16h bv;
            int nn = nb * 16 + col;
            int kbase = kb * 32 + hiHalf * 8;
            #pragma unroll
            for (int e = 0; e < 16; ++e)
                bv[e] = (_Float16)W2[nn * 64 + kbase + e + ((e >= 8) ? 8 : 0)];
            Bop[nb * 2 + kb] = bv;
        }
    }

    // ---- per-lane layer-3/bias constants ----
    float b2r[4], w3r[4];
    #pragma unroll
    for (int nb = 0; nb < 4; ++nb) {
        b2r[nb] = b2[nb * 16 + col];
        w3r[nb] = W3[nb * 16 + col];
    }

    const bool active = (lane < 8) || (lane >= 24);   // rows (lane&15) match

    for (int t = gwave; t < nTiles; t += nWaves) {
        int myrow = t * 16 + col;
        float xv = 0.0f, yv = 0.0f, Xv = 0.0f, Yv = 0.0f;
        if (myrow < n) {
            xv = x[myrow]; yv = y[myrow];
            Xv = xv + ex[myrow];
            Yv = yv + ey[myrow];
        }

        // ---- layer 1: packed f16 FMA + ReLU straight into A-operand layout ----
        v2h Xh = (v2h){(_Float16)Xv, (_Float16)Xv};
        v2h Yh = (v2h){(_Float16)Yv, (_Float16)Yv};
        v16h a0, a1;
        #pragma unroll
        for (int i = 0; i < 16; ++i) {
            v2h h = Xh * w1xp[i] + Yh * w1yp[i] + b1p[i];
            _Float16 z = (_Float16)0.0f;
            _Float16 h0 = (h[0] > z) ? h[0] : z;
            _Float16 h1v = (h[1] > z) ? h[1] : z;
            if (i < 8) { a0[2 * i] = h0; a0[2 * i + 1] = h1v; }
            else       { a1[2 * (i - 8)] = h0; a1[2 * (i - 8) + 1] = h1v; }
        }

        // ---- layer 2: 8x v_wmma_f32_16x16x32_f16 (K=64, N=64) ----
        v8f acc[4];
        #pragma unroll
        for (int nb = 0; nb < 4; ++nb) {
            v8f c0 = {};
            c0 = __builtin_amdgcn_wmma_f32_16x16x32_f16(
                     false, a0, false, Bop[nb * 2 + 0], (short)0, c0, false, false);
            acc[nb] = __builtin_amdgcn_wmma_f32_16x16x32_f16(
                     false, a1, false, Bop[nb * 2 + 1], (short)0, c0, false, false);
        }

        // ---- layer 3 partials: ReLU(acc + b2) * w3, per C-layout row r ----
        float s[8];
        #pragma unroll
        for (int r = 0; r < 8; ++r) s[r] = 0.0f;
        #pragma unroll
        for (int nb = 0; nb < 4; ++nb) {
            #pragma unroll
            for (int r = 0; r < 8; ++r) {
                float h2 = fmaxf(acc[nb][r] + b2r[nb], 0.0f);
                s[r] = fmaf(h2, w3r[nb], s[r]);
            }
        }

        // ---- reduce across the 16 lanes of each half ----
        #pragma unroll
        for (int m = 1; m <= 8; m <<= 1) {
            #pragma unroll
            for (int r = 0; r < 8; ++r)
                s[r] += __shfl_xor(s[r], m, 32);
        }

        // lanes 0-15 now hold out[0..7] in s[0..7]; lanes 16-31 hold out[8..15].
        int rsel = lane & 7;
        float u = s[0];
        u = (rsel == 1) ? s[1] : u;
        u = (rsel == 2) ? s[2] : u;
        u = (rsel == 3) ? s[3] : u;
        u = (rsel == 4) ? s[4] : u;
        u = (rsel == 5) ? s[5] : u;
        u = (rsel == 6) ? s[6] : u;
        u = (rsel == 7) ? s[7] : u;
        u -= dotT;

        if (active && myrow < n) {
            // Hill dynamics: S=0.5 -> s^2 = 0.25; A1=A2=1, B=0.2, K=1.1, scale=10
            float xs = xv * 0.1f, ys = yv * 0.1f;
            float x2 = xs * xs,  y2 = ys * ys;
            float invx = 1.0f / (0.25f + x2);
            float invy = 1.0f / (0.25f + y2);
            float hx = x2 * invx, hy = y2 * invy;
            float gx = 0.25f * invx, gy = 0.25f * invy;
            float dx = 10.0f * (fmaf(u, hx, hx) + 0.2f * gy - 1.1f * xs);
            float dy = 10.0f * (hy + 0.2f * gx - 1.1f * ys);
            out[myrow]         = dx;
            out[n + myrow]     = dy;
            out[2 * n + myrow] = -dx;
            out[3 * n + myrow] = -dy;
        }
    }
}

// ---------------------------------------------------------------------------
extern "C" void kernel_launch(void* const* d_in, const int* in_sizes, int n_in,
                              void* d_out, int out_size, void* d_ws, size_t ws_size,
                              hipStream_t stream) {
    const float* x  = (const float*)d_in[0];
    const float* y  = (const float*)d_in[1];
    const float* ex = (const float*)d_in[2];
    const float* ey = (const float*)d_in[3];
    const float* W1 = (const float*)d_in[4];
    const float* b1 = (const float*)d_in[5];
    const float* W2 = (const float*)d_in[6];
    const float* b2 = (const float*)d_in[7];
    const float* W3 = (const float*)d_in[8];
    float* out = (float*)d_out;
    float* ws  = (float*)d_ws;

    int n = in_sizes[0];
    int nTiles = (n + 15) / 16;

    netc_target_kernel<<<1, 64, 0, stream>>>(W1, b1, W2, b2, W3, ws);

    int blocks = (nTiles + 7) / 8;       // 8 waves per 256-thread block
    if (blocks > 1024) blocks = 1024;
    if (blocks < 1) blocks = 1;
    netc_main_kernel<<<blocks, 256, 0, stream>>>(x, y, ex, ey, W1, b1, W2, b2, W3,
                                                 ws, out, n, nTiles);
}
